// LASEREmbedderI_6150393168225
// MI455X (gfx1250) — matrix operand order repaired and checked by reading, not compiled, not run on hardware
//
#include <hip/hip_runtime.h>

typedef _Float16 v8h  __attribute__((ext_vector_type(8)));
typedef _Float16 v16h __attribute__((ext_vector_type(16)));
typedef float    v8f  __attribute__((ext_vector_type(8)));

#define TT 512
#define BB 32
#define EE 320
#define HH 512
#define LL 5

// ---- workspace layout (bytes) ----
#define NWI  18087936   // packed W_ih halves: 2*2048*320 + 8*2048*1024
#define NWH  10485760   // packed W_hh halves: 10*2048*512
#define NXE  5242880    // embedded x halves: 512*32*320
#define NHB  655360     // h double buffer halves: 2*5*32*1024
#define WS_WI ((size_t)0)
#define WS_WH ((size_t)(2*(size_t)NWI))                 // 36,175,872
#define WS_XE ((size_t)(WS_WH + 2*(size_t)NWH))        // 57,147,392
#define WS_HB ((size_t)(WS_XE + 2*(size_t)NXE))        // 67,633,152
#define WS_FL ((size_t)(WS_HB + 2*(size_t)NHB))        // 68,943,872
#define WS_NEED (WS_FL + 256)

__device__ __forceinline__ float sigm(float x) { return 1.0f / (1.0f + __expf(-x)); }

__device__ __forceinline__ void spin_ge(unsigned* p, unsigned tgt) {
  while (__hip_atomic_load(p, __ATOMIC_RELAXED, __HIP_MEMORY_SCOPE_AGENT) < tgt)
    __builtin_amdgcn_s_sleep(2);
}

__device__ __forceinline__ v16h cat16(v8h a, v8h b) {
  return __builtin_shufflevector(a, b, 0,1,2,3,4,5,6,7,8,9,10,11,12,13,14,15);
}

// One GEMM phase: acc += X[32 x 32*ktiles] * Wpacked, for this wave's 32 output cols.
// A layout (16-bit 16x32): lane<16 -> M=lane, K 0..7 & 16..23 ; lane>=16 -> K 8..15 & 24..31.
// B packed per (ktile, n): 32 contiguous halves over K; lanes 0..15 take K0..7 & 16..23, 16..31 take K8..15 & 24..31.
__device__ __forceinline__ void gemm_phase(
    const _Float16* __restrict__ xb, int xstride, int ktiles,
    const _Float16* __restrict__ wp, int ngl0, int mlo, int hisel,
    v8f& a00, v8f& a01, v8f& a10, v8f& a11)
{
  const int hi8 = hisel * 8;
  for (int kt = 0; kt < ktiles; ++kt) {
    const _Float16* pa0 = xb + mlo * xstride + kt * 32 + hi8;
    const _Float16* pa1 = pa0 + 16 * xstride;
    v16h A0 = cat16(*(const v8h*)pa0, *(const v8h*)(pa0 + 16));
    v16h A1 = cat16(*(const v8h*)pa1, *(const v8h*)(pa1 + 16));
    const _Float16* pb0 = wp + ((((size_t)kt << 11) + ngl0) << 5) + hisel * 8;
    const _Float16* pb1 = pb0 + (16 << 5);
    v16h B0 = cat16(*(const v8h*)pb0, *(const v8h*)(pb0 + 16));
    v16h B1 = cat16(*(const v8h*)pb1, *(const v8h*)(pb1 + 16));
    // Locality 3 -> WGP-scope prefetch: pulls next k-tile of B from L2 into the
    // near caches (SYS-scope prefetch would stop at GL2, which B already occupies).
    __builtin_prefetch(pb0 + ((size_t)2048 << 5), 0, 3);
    a00 = __builtin_amdgcn_wmma_f32_16x16x32_f16(false, A0, false, B0, (short)0, a00, false, false);
    a01 = __builtin_amdgcn_wmma_f32_16x16x32_f16(false, A0, false, B1, (short)0, a01, false, false);
    a10 = __builtin_amdgcn_wmma_f32_16x16x32_f16(false, A1, false, B0, (short)0, a10, false, false);
    a11 = __builtin_amdgcn_wmma_f32_16x16x32_f16(false, A1, false, B1, (short)0, a11, false, false);
  }
}

// ---------- prep kernels ----------
__global__ void k_pack_wi(const float* __restrict__ w0, const float* __restrict__ w1,
                          _Float16* __restrict__ wi) {
  int i = blockIdx.x * blockDim.x + threadIdx.x;
  if (i >= NWI) return;
  float v;
  if (i < 1310720) {                     // layer 0: [D][2048][320]
    int dd = i / 655360, rem = i % 655360;
    int kt = rem >> 16, n = (rem >> 5) & 2047, kk = rem & 31;
    v = w0[((size_t)dd * 2048 + n) * 320 + kt * 32 + kk];
  } else {                               // layers 1..4: [8][2048][1024]
    int p = i - 1310720;
    int cc = p / 2097152, rem = p % 2097152;
    int kt = rem >> 16, n = (rem >> 5) & 2047, kk = rem & 31;
    v = w1[((size_t)cc * 2048 + n) * 1024 + kt * 32 + kk];
  }
  wi[i] = (_Float16)v;
}

__global__ void k_pack_wh(const float* __restrict__ ws, _Float16* __restrict__ wh) {
  int i = blockIdx.x * blockDim.x + threadIdx.x;
  if (i >= NWH) return;
  int cc = i >> 20, rem = i & 1048575;
  int kt = rem >> 16, n = (rem >> 5) & 2047, kk = rem & 31;
  wh[i] = (_Float16)ws[((size_t)cc * 2048 + n) * 512 + kt * 32 + kk];
}

__global__ void k_embed(const int* __restrict__ tok, const float* __restrict__ emb,
                        _Float16* __restrict__ xe) {
  int i = blockIdx.x * blockDim.x + threadIdx.x;
  if (i >= NXE) return;
  int e = i % 320, tb = i / 320;
  int b = tb & 31, t = tb >> 5;          // xe layout [T][B][E]
  int tk = tok[(size_t)b * TT + t];      // tokens are [B][T]
  xe[i] = (_Float16)emb[(size_t)tk * 320 + e];
}

// ---------- persistent LSTM kernel ----------
// 40 blocks = 10 cells (l,d) x 4 column-chunks (128 h-cols each). 512 threads = 16 waves.
// Wave w: gate q=w>>2, 32 gate-cols. Credit-flag pipeline over T=512 steps.
__global__ __launch_bounds__(512, 1) void lstm_persistent(
    const _Float16* __restrict__ wi, const _Float16* __restrict__ wh,
    const float* __restrict__ bias, const _Float16* __restrict__ xe,
    _Float16* __restrict__ hbuf, unsigned* flags, float* __restrict__ out)
{
  const int tid = threadIdx.x;
  const int bid = blockIdx.x;
  const int cid = bid >> 2;              // cell 0..9 (= l*2+d)
  const int c   = bid & 3;               // 128-wide h chunk
  const int l   = cid >> 1, d = cid & 1;
  const int w    = tid >> 5;
  const int lane = tid & 31;
  const int q  = w >> 2;                 // gate: 0=i 1=f 2=g 3=o
  const int nb = w & 3;
  const int mlo = lane & 15, hisel = lane >> 4;
  const int ncol = c * 128 + nb * 32 + (lane & 15);   // col within gate [0,512)
  const int ngl0 = q * 512 + ncol;                    // col within 4H  [0,2048)
  const int Kx = (l == 0) ? EE : 2 * HH;
  const _Float16* wic = wi + ((cid < 2) ? (size_t)cid * 655360
                                        : 1310720 + (size_t)(cid - 2) * 2097152);
  const _Float16* whc = wh + (size_t)cid * 1048576;
  const float bv0 = bias[cid * 2048 + ngl0];
  const float bv1 = bias[cid * 2048 + ngl0 + 16];

  __shared__ float lds_g[4][32][128];    // staged gates (f32)
  __shared__ float lds_c[32][128];       // persistent cell state
  for (int i = tid; i < 32 * 128; i += 512) (&lds_c[0][0])[i] = 0.0f;
  __syncthreads();

  for (int t = 0; t < TT; ++t) {
    if (tid == 0) {
      if (l > 0) {                        // input h(t) of layer l-1 ready (both dirs)
        spin_ge(&flags[2 * l - 2], 4u * (unsigned)(t + 1));
        spin_ge(&flags[2 * l - 1], 4u * (unsigned)(t + 1));
      }
      if (t >= 1)                         // own-cell h(t-1) fully written by siblings
        spin_ge(&flags[cid], 4u * (unsigned)t);
      if (t >= 2) {                       // anti-dep: consumers finished reading h(t-2)
        if (l < 4) {
          spin_ge(&flags[2 * l + 2], 4u * (unsigned)(t - 1));
          spin_ge(&flags[2 * l + 3], 4u * (unsigned)(t - 1));
        } else if (d == 0) {
          spin_ge(&flags[9], 4u * (unsigned)(t - 1));
        }
      }
    }
    __syncthreads();
    __threadfence();                      // acquire peer h writes

    const _Float16* xb; int xstride;
    if (l == 0) { xb = xe + (size_t)t * BB * EE; xstride = EE; }
    else        { xb = hbuf + (size_t)((t & 1) * LL + (l - 1)) * BB * 1024; xstride = 1024; }
    const _Float16* hb = hbuf + (size_t)(((t + 1) & 1) * LL + l) * BB * 1024 + d * HH;

    v8f a00 = {bv0, bv0, bv0, bv0, bv0, bv0, bv0, bv0};
    v8f a01 = {bv1, bv1, bv1, bv1, bv1, bv1, bv1, bv1};
    v8f a10 = a00, a11 = a01;
    gemm_phase(xb, xstride, Kx >> 5, wic, ngl0, mlo, hisel, a00, a01, a10, a11);
    gemm_phase(hb, 1024,   HH >> 5, whc, ngl0, mlo, hisel, a00, a01, a10, a11);

    // D layout: lane<16 -> M=reg, N=lane ; lane>=16 -> M=8+reg, N=lane-16
    const int colb = nb * 32 + (lane & 15);
    #pragma unroll
    for (int r = 0; r < 8; ++r) {
      int b0 = hisel * 8 + r, b1 = 16 + b0;
      lds_g[q][b0][colb     ] = a00[r];
      lds_g[q][b0][colb + 16] = a01[r];
      lds_g[q][b1][colb     ] = a10[r];
      lds_g[q][b1][colb + 16] = a11[r];
    }
    __syncthreads();

    _Float16* hout = hbuf + (size_t)((t & 1) * LL + l) * BB * 1024 + d * HH + c * 128;
    #pragma unroll
    for (int i = 0; i < 8; ++i) {
      int e = tid + i * 512;
      int b_ = e >> 7, jj = e & 127;
      float gi = lds_g[0][b_][jj], gf = lds_g[1][b_][jj];
      float gg = lds_g[2][b_][jj], go = lds_g[3][b_][jj];
      float cp = lds_c[b_][jj];
      float cn = sigm(gf) * cp + sigm(gi) * tanhf(gg);
      lds_c[b_][jj] = cn;
      hout[(size_t)b_ * 1024 + jj] = (_Float16)(sigm(go) * tanhf(cn));
    }
    __threadfence();                      // release h stores
    __syncthreads();

    if (cid == 9) {                       // (l=4,d=1): direction-max -> output
      if (tid == 0) spin_ge(&flags[8], 4u * (unsigned)(t + 1));
      __syncthreads();
      __threadfence();
      const _Float16* h4 = hbuf + (size_t)((t & 1) * LL + 4) * BB * 1024;
      #pragma unroll
      for (int i = 0; i < 8; ++i) {
        int e = tid + i * 512;
        int b_ = e >> 7, jj = e & 127;
        int j = c * 128 + jj;
        float h0 = (float)h4[(size_t)b_ * 1024 + j];
        float h1 = (float)h4[(size_t)b_ * 1024 + 512 + j];
        out[((size_t)t * BB + b_) * HH + j] = fmaxf(h0, h1);
      }
      __syncthreads();                    // all output reads done before signaling
    }
    if (tid == 0) atomicAdd(&flags[cid], 1u);
  }
}

extern "C" void kernel_launch(void* const* d_in, const int* in_sizes, int n_in,
                              void* d_out, int out_size, void* d_ws, size_t ws_size,
                              hipStream_t stream) {
  if (ws_size < WS_NEED) return;  // fail loudly (d_out stays poisoned)
  const int*   tokens = (const int*)d_in[0];
  const float* emb    = (const float*)d_in[1];
  const float* W_ih0  = (const float*)d_in[2];
  const float* W_ih   = (const float*)d_in[3];
  const float* W_hh   = (const float*)d_in[4];
  const float* bias   = (const float*)d_in[5];
  char* ws = (char*)d_ws;
  _Float16* wi = (_Float16*)(ws + WS_WI);
  _Float16* wh = (_Float16*)(ws + WS_WH);
  _Float16* xe = (_Float16*)(ws + WS_XE);
  _Float16* hb = (_Float16*)(ws + WS_HB);
  unsigned* fl = (unsigned*)(ws + WS_FL);

  hipMemsetAsync(ws + WS_HB, 0, 2 * (size_t)NHB + 64, stream);  // h buffers + flags
  k_pack_wi<<<(NWI + 255) / 256, 256, 0, stream>>>(W_ih0, W_ih, wi);
  k_pack_wh<<<(NWH + 255) / 256, 256, 0, stream>>>(W_hh, wh);
  k_embed  <<<(NXE + 255) / 256, 256, 0, stream>>>(tokens, emb, xe);
  lstm_persistent<<<40, 512, 0, stream>>>(wi, wh, bias, xe, hb, fl, (float*)d_out);
}
